// MyGIN_86036784873977
// MI455X (gfx1250) — matrix operand order repaired and checked by reading, not compiled, not run on hardware
//
#include <hip/hip_runtime.h>
#include <hip/hip_bf16.h>
#include <math.h>

#define NN  50000
#define EE  600000
#define GG  1024
#define HD  128
#define FIN 74
#define LL  5
#define CC  2

typedef __attribute__((ext_vector_type(16))) __bf16 v16bf;
typedef __attribute__((ext_vector_type(8)))  float  v8f;

__device__ __forceinline__ float sigmoidf_(float v) { return 1.f / (1.f + __expf(-v)); }

// ---------------- WMMA fragment helpers (A/B in contiguous k = 16h + s order; C/D per CDNA5 ISA 7.12.2) ----------------

// A: 16x32 bf16 tile from LDS, row-major with leading dim 384.
// lanes 0-15 : row=lane, elems 0..15 -> K 0..15
// lanes 16-31: row=lane-16, elems 0..15 -> K 16..31
__device__ __forceinline__ v16bf load_a_frag(const __bf16* sA, int klo) {
  const int lane = threadIdx.x & 31;
  const int row  = lane & 15;
  const int kh   = (lane >> 4) * 16;
  const __bf16* p = sA + row * 384 + klo;
  v16bf a;
#pragma unroll
  for (int e = 0; e < 16; ++e) {
    int k = e + kh;
    a[e] = p[k];
  }
  return a;
}

// B: 32x16 bf16 tile. Weights stored [col][k] (row-major, ldk = K of GEMM).
// lanes 0-15: col=lane, K = klo..klo+15 ; lanes 16-31: col=lane-16, K = klo+16..klo+31
__device__ __forceinline__ v16bf load_b_frag(const __bf16* W, int colbase, int klo, int ldk) {
  const int lane = threadIdx.x & 31;
  const __bf16* p = W + (size_t)(colbase + (lane & 15)) * ldk + klo + ((lane >> 4) << 4);
  return *(const v16bf*)p;   // 32B aligned: ldk in {128,256}, klo multiple of 32
}

__device__ __forceinline__ v8f wmma_bf16(v16bf a, v16bf b, v8f c) {
  return __builtin_amdgcn_wmma_f32_16x16x32_bf16(false, a, false, b, (short)0, c, false, false);
}

// ---------------- small prep kernels ----------------

__global__ void k_cvt_w(const float* __restrict__ wih, const float* __restrict__ whh,
                        __bf16* __restrict__ wihB, __bf16* __restrict__ whhB) {
  int i = blockIdx.x * blockDim.x + threadIdx.x;
  if (i < 384 * 256) wihB[i] = (__bf16)wih[i];
  if (i < 384 * 128) whhB[i] = (__bf16)whh[i];
}

__global__ void k_inproj(const float* __restrict__ feats, const float* __restrict__ Win,
                         const float* __restrict__ bin, float* __restrict__ h) {
  long tid = (long)blockIdx.x * 256 + threadIdx.x;
  int n = (int)(tid >> 7), f = (int)(tid & 127);
  if (n >= NN) return;
  float s = bin[f];
  const float* fr = feats + (size_t)n * FIN;
#pragma unroll 2
  for (int k = 0; k < FIN; ++k) s += fr[k] * Win[k * HD + f];
  h[(size_t)n * HD + f] = s;
}

__global__ void k_bn_pack(const float* __restrict__ h, const float* __restrict__ gamma,
                          const float* __restrict__ beta, const float* __restrict__ mean,
                          const float* __restrict__ var, float* __restrict__ x,
                          __bf16* __restrict__ Acat, float* __restrict__ agg, int l) {
  long tid = (long)blockIdx.x * 256 + threadIdx.x;
  int n = (int)(tid >> 7), f = (int)(tid & 127);
  if (n >= NN) return;
  int pf = l * HD + f;
  size_t idx = (size_t)n * HD + f;
  float xv = (h[idx] - mean[pf]) * rsqrtf(var[pf] + 1e-5f) * gamma[pf] + beta[pf];
  x[idx] = xv;
  __bf16* ar = Acat + (size_t)n * 384;
  ar[f]       = (__bf16)(1.00001f * xv);  // (1 + gin_eps) * x
  ar[256 + f] = (__bf16)xv;               // GRU hidden input
  agg[idx] = 0.f;
}

__global__ void k_scatter(const float* __restrict__ x, const int* __restrict__ src,
                          const int* __restrict__ dst, float* __restrict__ agg) {
  long tid = (long)blockIdx.x * 256 + threadIdx.x;
  if (tid >= (long)EE * 32) return;
  int e  = (int)(tid >> 5);
  int f4 = (int)(tid & 31) << 2;
  int s = src[e], d = dst[e];
  const float4 v = *(const float4*)(x + (size_t)s * HD + f4);
  float* a = agg + (size_t)d * HD + f4;
  atomicAdd(a + 0, v.x); atomicAdd(a + 1, v.y);
  atomicAdd(a + 2, v.z); atomicAdd(a + 3, v.w);
}

__global__ void k_agg_pack(const float* __restrict__ agg, __bf16* __restrict__ Acat) {
  long tid = (long)blockIdx.x * 256 + threadIdx.x;
  int n = (int)(tid >> 7), f = (int)(tid & 127);
  if (n >= NN) return;
  Acat[(size_t)n * 384 + 128 + f] = (__bf16)agg[(size_t)n * HD + f];
}

__global__ void k_nodew(const float* __restrict__ h, const float* __restrict__ waw,
                        const float* __restrict__ baw, float* __restrict__ wnode) {
  int lane = threadIdx.x & 31;
  int n = blockIdx.x * 4 + (threadIdx.x >> 5);
  if (n >= NN) return;
  const float* hp = h + (size_t)n * HD;
  float s = 0.f;
#pragma unroll
  for (int i = 0; i < 4; ++i) s += hp[lane + 32 * i] * waw[lane + 32 * i];
#pragma unroll
  for (int off = 16; off; off >>= 1) s += __shfl_xor(s, off, 32);
  if (lane == 0) wnode[n] = sigmoidf_(s + baw[0]);
}

// block-per-graph segmented sum/max (node2graph is sorted)
__global__ void k_readout(const float* __restrict__ h, const float* __restrict__ wnode,
                          const int* __restrict__ n2g, float* __restrict__ rep, int l) {
  int g = blockIdx.x, f = threadIdx.x;
  int lo = 0, hi = NN;
  while (lo < hi) { int m = (lo + hi) >> 1; if (n2g[m] < g) lo = m + 1; else hi = m; }
  int beg = lo;
  hi = NN;
  while (lo < hi) { int m = (lo + hi) >> 1; if (n2g[m] < g + 1) lo = m + 1; else hi = m; }
  int end = lo;
  float ws = 0.f, hm = -INFINITY;
  for (int n = beg; n < end; ++n) {
    float hv = h[(size_t)n * HD + f];
    ws += wnode[n] * hv;
    hm = fmaxf(hm, hv);
  }
  if (beg == end) hm = 0.f;   // where(isfinite) on empty segment
  float* r = rep + ((size_t)l * GG + g) * (2 * HD);
  r[f] = ws; r[HD + f] = hm;
}

// ---------------- fused GRU: bf16 WMMA GEMMs + gates + CELU + residual ----------------
// block = 32 nodes (two 16-row M-subtiles); 8 waves, one 16-column tile of Hd each.
// Each B fragment feeds two WMMAs -> 2x weight (L2) reuse vs one subtile per block.
__global__ void __launch_bounds__(256) k_gru(const __bf16* __restrict__ Acat,
                                             const __bf16* __restrict__ Wih,
                                             const __bf16* __restrict__ Whh,
                                             const float* __restrict__ bih,
                                             const float* __restrict__ bhh,
                                             const float* __restrict__ x,
                                             float* __restrict__ h) {
  __shared__ __bf16 sA[32 * 384];          // 24 KB
  const int mbase = blockIdx.x * 32;
  const int t = threadIdx.x;
  __builtin_prefetch(Wih, 0, 3);
  __builtin_prefetch(Whh, 0, 3);

  // Stage up-to 32x384 bf16 A-tile (24 KB) via CDNA5 async global->LDS copies.
  const int rows = (NN - mbase) < 32 ? (NN - mbase) : 32;
  const uint32_t validBytes = (uint32_t)rows * 384u * 2u;
  const char* gsrc = (const char*)(Acat + (size_t)mbase * 384);
  uint32_t lds0 = (uint32_t)(uintptr_t)(&sA[0]);
#pragma unroll
  for (int i = 0; i < 6; ++i) {
    uint32_t byte = (uint32_t)(t + i * 256) * 16u;
    if (byte < validBytes) {
      asm volatile("global_load_async_to_lds_b128 %0, %1, off"
                   :: "v"(lds0 + byte), "v"(gsrc + byte) : "memory");
    }
  }
  asm volatile("s_wait_asynccnt 0" ::: "memory");
  __syncthreads();

  const int lane = t & 31;
  const int colbase = (t >> 5) * 16;

  v8f acc[2][6];                    // [subtile][0..2: gi r,z,n ; 3..5: gh r,z,n]
#pragma unroll
  for (int s = 0; s < 2; ++s)
#pragma unroll
    for (int i = 0; i < 6; ++i)
#pragma unroll
      for (int e = 0; e < 8; ++e) acc[s][i][e] = 0.f;

  // gi = [(1+eps)x | agg] @ Wih^T   (K = 256)
#pragma unroll
  for (int kk = 0; kk < 256; kk += 32) {
    v16bf a0 = load_a_frag(sA, kk);
    v16bf a1 = load_a_frag(sA + 16 * 384, kk);
#pragma unroll
    for (int gch = 0; gch < 3; ++gch) {
      v16bf b = load_b_frag(Wih, gch * HD + colbase, kk, 256);
      acc[0][gch] = wmma_bf16(a0, b, acc[0][gch]);
      acc[1][gch] = wmma_bf16(a1, b, acc[1][gch]);
    }
  }
  // gh = x @ Whh^T   (K = 128; x lives in A cols 256..383)
#pragma unroll
  for (int kk = 0; kk < 128; kk += 32) {
    v16bf a0 = load_a_frag(sA, 256 + kk);
    v16bf a1 = load_a_frag(sA + 16 * 384, 256 + kk);
#pragma unroll
    for (int gch = 0; gch < 3; ++gch) {
      v16bf b = load_b_frag(Whh, gch * HD + colbase, kk, 128);
      acc[0][3 + gch] = wmma_bf16(a0, b, acc[0][3 + gch]);
      acc[1][3 + gch] = wmma_bf16(a1, b, acc[1][3 + gch]);
    }
  }

  // C/D layout: elem v -> row = v + 8*(lane>=16), col = lane%16
  const int hiHalf = lane >> 4;
  const int col = colbase + (lane & 15);
  const float bir = bih[col], biz = bih[HD + col], bin = bih[2 * HD + col];
  const float bhr = bhh[col], bhz = bhh[HD + col], bhn = bhh[2 * HD + col];
#pragma unroll
  for (int s = 0; s < 2; ++s) {
#pragma unroll
    for (int v = 0; v < 8; ++v) {
      int node = mbase + s * 16 + v + 8 * hiHalf;
      if (node >= NN) continue;
      size_t idx = (size_t)node * HD + col;
      float xv = x[idx];
      float r  = sigmoidf_(acc[s][0][v] + bir + acc[s][3][v] + bhr);
      float z  = sigmoidf_(acc[s][1][v] + biz + acc[s][4][v] + bhz);
      float nn = tanhf(acc[s][2][v] + bin + r * (acc[s][5][v] + bhn));
      float hc = (1.f - z) * nn + z * xv;
      hc = hc > 0.f ? hc : (__expf(hc) - 1.f);   // celu, alpha=1
      h[idx] += hc;                               // residual
    }
  }
}

// ---------------- BiLSTM head: y_f[0] needs 1 fwd step, y_b[0] needs full bwd scan ----
__global__ void k_lstm(const float* __restrict__ rep,
                       const float* __restrict__ wihf, const float* __restrict__ whhf,
                       const float* __restrict__ bihf, const float* __restrict__ bhhf,
                       const float* __restrict__ wihb, const float* __restrict__ whhb,
                       const float* __restrict__ bihb, const float* __restrict__ bhhb,
                       const float* __restrict__ Wout, const float* __restrict__ bout,
                       float* __restrict__ out) {
  const int g = blockIdx.x, j = threadIdx.x;  // 128 threads
  __shared__ float sx[2 * HD];
  __shared__ float sh[HD];
  __shared__ float scat[2 * HD];

  for (int k = j; k < 2 * HD; k += HD) sx[k] = rep[(size_t)g * 2 * HD + k];
  __syncthreads();
  {
    float gi = bihf[j] + bhhf[j];
    float gf = bihf[HD + j] + bhhf[HD + j];
    float gg = bihf[2 * HD + j] + bhhf[2 * HD + j];
    float go = bihf[3 * HD + j] + bhhf[3 * HD + j];
    for (int k = 0; k < 2 * HD; ++k) {
      float xv = sx[k];
      gi += xv * wihf[(size_t)j * 2 * HD + k];
      gf += xv * wihf[(size_t)(HD + j) * 2 * HD + k];
      gg += xv * wihf[(size_t)(2 * HD + j) * 2 * HD + k];
      go += xv * wihf[(size_t)(3 * HD + j) * 2 * HD + k];
    }
    (void)gf; // h0=c0=0 -> f-gate term vanishes
    float c = sigmoidf_(gi) * tanhf(gg);
    scat[j] = sigmoidf_(go) * tanhf(c);        // y_f[0]
  }
  float cb = 0.f;
  __syncthreads();
  sh[j] = 0.f;
  __syncthreads();
  for (int tt = LL - 1; tt >= 0; --tt) {
    for (int k = j; k < 2 * HD; k += HD)
      sx[k] = rep[((size_t)tt * GG + g) * 2 * HD + k];
    __syncthreads();
    float gi = bihb[j] + bhhb[j];
    float gf = bihb[HD + j] + bhhb[HD + j];
    float gg = bihb[2 * HD + j] + bhhb[2 * HD + j];
    float go = bihb[3 * HD + j] + bhhb[3 * HD + j];
    for (int k = 0; k < 2 * HD; ++k) {
      float xv = sx[k];
      gi += xv * wihb[(size_t)j * 2 * HD + k];
      gf += xv * wihb[(size_t)(HD + j) * 2 * HD + k];
      gg += xv * wihb[(size_t)(2 * HD + j) * 2 * HD + k];
      go += xv * wihb[(size_t)(3 * HD + j) * 2 * HD + k];
    }
    for (int k = 0; k < HD; ++k) {
      float hv = sh[k];
      gi += hv * whhb[(size_t)j * HD + k];
      gf += hv * whhb[(size_t)(HD + j) * HD + k];
      gg += hv * whhb[(size_t)(2 * HD + j) * HD + k];
      go += hv * whhb[(size_t)(3 * HD + j) * HD + k];
    }
    cb = sigmoidf_(gf) * cb + sigmoidf_(gi) * tanhf(gg);
    float hn = sigmoidf_(go) * tanhf(cb);
    __syncthreads();
    sh[j] = hn;
    __syncthreads();
  }
  scat[HD + j] = sh[j];                        // y_b[0]
  __syncthreads();
  if (j < CC) {
    float s = bout[j];
    for (int k = 0; k < 2 * HD; ++k) s += scat[k] * Wout[k * CC + j];
    out[(size_t)g * CC + j] = s;
  }
}

// ---------------- host orchestration ----------------

extern "C" void kernel_launch(void* const* d_in, const int* in_sizes, int n_in,
                              void* d_out, int out_size, void* d_ws, size_t ws_size,
                              hipStream_t stream) {
  (void)in_sizes; (void)n_in; (void)out_size; (void)ws_size;
  const float* feats = (const float*)d_in[0];
  const int*   src   = (const int*)d_in[1];
  const int*   dst   = (const int*)d_in[2];
  const int*   n2g   = (const int*)d_in[3];
  const float* Win   = (const float*)d_in[4];
  const float* bin   = (const float*)d_in[5];
  const float* bng   = (const float*)d_in[6];
  const float* bnb   = (const float*)d_in[7];
  const float* bnm   = (const float*)d_in[8];
  const float* bnv   = (const float*)d_in[9];
  const float* gwih  = (const float*)d_in[10];
  const float* gwhh  = (const float*)d_in[11];
  const float* gbih  = (const float*)d_in[12];
  const float* gbhh  = (const float*)d_in[13];
  const float* waw   = (const float*)d_in[14];
  const float* baw   = (const float*)d_in[15];
  const float* wihf  = (const float*)d_in[16];
  const float* whhf  = (const float*)d_in[17];
  const float* bihf  = (const float*)d_in[18];
  const float* bhhf  = (const float*)d_in[19];
  const float* wihb  = (const float*)d_in[20];
  const float* whhb  = (const float*)d_in[21];
  const float* bihb  = (const float*)d_in[22];
  const float* bhhb  = (const float*)d_in[23];
  const float* Wout  = (const float*)d_in[24];
  const float* bout  = (const float*)d_in[25];

  char* ws = (char*)d_ws;
  size_t off = 0;
  auto alloc = [&](size_t bytes) -> void* {
    void* p = ws + off;
    off = (off + bytes + 255) & ~(size_t)255;
    return p;
  };
  float*  h     = (float*)alloc((size_t)NN * HD * 4);
  float*  x     = (float*)alloc((size_t)NN * HD * 4);
  float*  agg   = (float*)alloc((size_t)NN * HD * 4);
  __bf16* Acat  = (__bf16*)alloc((size_t)NN * 384 * 2);
  float*  wnode = (float*)alloc((size_t)NN * 4);
  float*  rep   = (float*)alloc((size_t)LL * GG * 2 * HD * 4);
  __bf16* WihB  = (__bf16*)alloc((size_t)384 * 256 * 2);
  __bf16* WhhB  = (__bf16*)alloc((size_t)384 * 128 * 2);

  k_cvt_w<<<384, 256, 0, stream>>>(gwih, gwhh, WihB, WhhB);
  k_inproj<<<(NN * HD) / 256, 256, 0, stream>>>(feats, Win, bin, h);

  for (int l = 0; l < LL; ++l) {
    k_bn_pack<<<(NN * HD) / 256, 256, 0, stream>>>(h, bng, bnb, bnm, bnv, x, Acat, agg, l);
    k_scatter<<<(EE * 32) / 256, 256, 0, stream>>>(x, src, dst, agg);
    k_agg_pack<<<(NN * HD) / 256, 256, 0, stream>>>(agg, Acat);
    k_nodew<<<NN / 4, 128, 0, stream>>>(h, waw, baw, wnode);
    k_readout<<<GG, HD, 0, stream>>>(h, wnode, n2g, rep, l);
    k_gru<<<(NN + 31) / 32, 256, 0, stream>>>(Acat, WihB, WhhB, gbih, gbhh, x, h);
  }

  k_lstm<<<GG, HD, 0, stream>>>(rep, wihf, whhf, bihf, bhhf,
                                wihb, whhb, bihb, bhhb, Wout, bout, (float*)d_out);
}
